// GNNReadabilityGAT_86260123173012
// MI455X (gfx1250) — matrix operand order repaired and checked by reading, not compiled, not run on hardware
//
#include <hip/hip_runtime.h>
#include <hip/hip_bf16.h>

// ---------------------------------------------------------------------------
// GNN Readability GAT forward for MI455X (gfx1250, wave32, WMMA)
// ---------------------------------------------------------------------------

typedef __attribute__((ext_vector_type(16))) _Float16 v16h;
typedef __attribute__((ext_vector_type(8)))  _Float16 v8h;
typedef __attribute__((ext_vector_type(8)))  float    v8f;

#define F_IN     300
#define F_IN_PAD 320      /* padded to multiple of 32 */
#define HEADS    4
#define DIM      256
#define NCLS     6

// ---------------------------------------------------------------------------
// f32 -> f16 conversion with zero padding of the K dimension.
// src: [rows, K] f32 row-major      dst: [rows, Kpad] f16 row-major
// ---------------------------------------------------------------------------
__global__ void cvt_f16_pad(const float* __restrict__ src, _Float16* __restrict__ dst,
                            int rows, int K, int Kpad)
{
    const size_t i = (size_t)blockIdx.x * blockDim.x + threadIdx.x;
    if (i >= (size_t)rows * Kpad) return;
    const int r = (int)(i / Kpad);
    const int k = (int)(i % Kpad);
    dst[i] = (k < K) ? (_Float16)src[(size_t)r * K + k] : (_Float16)0.0f;
}

// ---------------------------------------------------------------------------
// WMMA GEMM: C[M,N] = A[M,Kpad] * B[N,Kpad]^T (+bias, optional ReLU)
// A, B are f16 row-major with Kpad % 32 == 0 (zero padded) -> no masking.
// Requires M % 16 == 0 and N % 64 == 0.
// One wave computes a 16(M) x 64(N) tile: 4 f32 accumulators, A frag reused.
// Block = 256 threads = 8 waves stacked along M -> 128 x 64 block tile.
//
// Fragment layouts used here (16-bit, wave32; A/B in contiguous k order, not the ISA table order):
//   A  lane L: row M = L&15; halfword h   -> K = k0 + 16*(L>=16) + h
//   B  lane L: col N = L&15; halfword h   -> K = k0 + 16*(L>=16) + h
//   C  lane L: col N = L&15; element i    -> row = i + 8*(L>=16)
// ---------------------------------------------------------------------------
__global__ __launch_bounds__(256)
void wmma_gemm_f16(const _Float16* __restrict__ A, const _Float16* __restrict__ B,
                   const float* __restrict__ bias, float* __restrict__ C,
                   int M, int N, int Kpad, int act /*0=none, 1=relu*/)
{
    const int lane    = threadIdx.x & 31;
    const int wave    = threadIdx.x >> 5;
    const int halfsel = lane >> 4;
    const int l15     = lane & 15;

    const int m0 = (blockIdx.x * 8 + wave) * 16;
    const int n0 = blockIdx.y * 64;
    if (m0 >= M) return;

    v8f acc[4] = {};

    const _Float16* __restrict__ Arow  = A + (size_t)(m0 + l15) * Kpad;
    const _Float16* __restrict__ Brow0 = B + (size_t)(n0 + l15) * Kpad;
    const size_t bstride = (size_t)16 * Kpad;   // 16 output cols per sub-tile

    for (int k0 = 0; k0 < Kpad; k0 += 32) {
        // A fragment: two 16-byte contiguous loads per lane
        const int ka = k0 + (halfsel << 4);
        const v8h alo = *(const v8h*)(Arow + ka);
        const v8h ahi = *(const v8h*)(Arow + ka + 8);
        const v16h afrag = __builtin_shufflevector(
            alo, ahi, 0, 1, 2, 3, 4, 5, 6, 7, 8, 9, 10, 11, 12, 13, 14, 15);

        // 4 B fragments: one 32-byte contiguous load each
        const int kb = k0 + (halfsel << 4);
#pragma unroll
        for (int s = 0; s < 4; ++s) {
            const v16h bfrag = *(const v16h*)(Brow0 + (size_t)s * bstride + kb);
            acc[s] = __builtin_amdgcn_wmma_f32_16x16x32_f16(
                false, afrag, false, bfrag, (short)0, acc[s], false, false);
        }
    }

#pragma unroll
    for (int s = 0; s < 4; ++s) {
        const int ncol = n0 + s * 16 + l15;
        const float bv = bias ? bias[ncol] : 0.0f;
#pragma unroll
        for (int i = 0; i < 8; ++i) {
            const int row = m0 + i + (halfsel << 3);
            float v = acc[s][i] + bv;
            if (act == 1) v = v > 0.0f ? v : 0.0f;
            C[(size_t)row * N + ncol] = v;
        }
    }
}

// ---------------------------------------------------------------------------
// attention projections: as[n,h] = <h[n,h,:], a_src[h,:]>, ad likewise.
// grid (N, heads), block = 32 (one wave); wave32 shfl reduction.
// ---------------------------------------------------------------------------
__global__ __launch_bounds__(32)
void alpha_kernel(const float* __restrict__ h, const float* __restrict__ a_src,
                  const float* __restrict__ a_dst, float* __restrict__ as,
                  float* __restrict__ ad, int heads, int D)
{
    const int n    = blockIdx.x;
    const int hd   = blockIdx.y;
    const int lane = threadIdx.x;
    const float* __restrict__ hp  = h + (size_t)n * heads * D + (size_t)hd * D;
    const float* __restrict__ wsv = a_src + (size_t)hd * D;
    const float* __restrict__ wdv = a_dst + (size_t)hd * D;
    float s = 0.0f, d = 0.0f;
    for (int i = lane; i < D; i += 32) {
        const float v = hp[i];
        s += v * wsv[i];
        d += v * wdv[i];
    }
#pragma unroll
    for (int off = 16; off > 0; off >>= 1) {
        s += __shfl_down(s, off, 32);
        d += __shfl_down(d, off, 32);
    }
    if (lane == 0) {
        as[n * heads + hd] = s;
        ad[n * heads + hd] = d;
    }
}

// ---------------------------------------------------------------------------
// monotone uint <-> float order encoding for atomic float max
// ---------------------------------------------------------------------------
__device__ __forceinline__ unsigned f2ord(float f) {
    unsigned u = __float_as_uint(f);
    return (u & 0x80000000u) ? ~u : (u | 0x80000000u);
}
__device__ __forceinline__ float ord2f(unsigned u) {
    u = (u & 0x80000000u) ? (u & 0x7FFFFFFFu) : ~u;
    return __uint_as_float(u);
}
#define NEG_INF_ORD 0x007FFFFFu   /* f2ord(-inf) */

__device__ __forceinline__ void edge_sd(const int* __restrict__ ei, int E, int e,
                                        int& s, int& d) {
    if (e < E) { s = ei[e]; d = ei[E + e]; }
    else       { s = e - E; d = s; }         // self loops appended
}

__global__ void fill_u32(unsigned* __restrict__ p, unsigned v, size_t n) {
    const size_t i = (size_t)blockIdx.x * blockDim.x + threadIdx.x;
    if (i < n) p[i] = v;
}

// per-edge logits + segment max (into ordered-uint buffer)
__global__ void edge_logits(const int* __restrict__ ei, int E, int tot, int heads,
                            const float* __restrict__ as, const float* __restrict__ ad,
                            float* __restrict__ ebuf, unsigned* __restrict__ emax)
{
    const int e = blockIdx.x * blockDim.x + threadIdx.x;
    if (e >= tot) return;
    int s, d; edge_sd(ei, E, e, s, d);
    for (int h = 0; h < heads; ++h) {
        float v = as[s * heads + h] + ad[d * heads + h];
        v = v > 0.0f ? v : 0.2f * v;                 // leaky relu 0.2
        ebuf[(size_t)e * heads + h] = v;
        atomicMax(&emax[d * heads + h], f2ord(v));
    }
}

// exp(e - max) + segment sum
__global__ void edge_softmax_num(const int* __restrict__ ei, int E, int tot, int heads,
                                 const unsigned* __restrict__ emax,
                                 float* __restrict__ ebuf, float* __restrict__ den)
{
    const int e = blockIdx.x * blockDim.x + threadIdx.x;
    if (e >= tot) return;
    int s, d; edge_sd(ei, E, e, s, d);
    (void)s;
    for (int h = 0; h < heads; ++h) {
        const float mx = ord2f(emax[d * heads + h]);
        const float ee = __expf(ebuf[(size_t)e * heads + h] - mx);
        ebuf[(size_t)e * heads + h] = ee;
        atomicAdd(&den[d * heads + h], ee);
    }
}

// weighted feature scatter: out[dst] += alpha * h[src]. one block per edge.
__global__ __launch_bounds__(256)
void edge_scatter(const int* __restrict__ ei, int E, int heads, int D,
                  const float* __restrict__ h, const float* __restrict__ ebuf,
                  const float* __restrict__ den, float* __restrict__ out)
{
    const int e = blockIdx.x;
    int s, d; edge_sd(ei, E, e, s, d);
    const int HD = heads * D;
    for (int f = threadIdx.x; f < HD; f += blockDim.x) {
        const int hh = f >> 8;                         // D == 256
        const float alpha = ebuf[(size_t)e * heads + hh] /
                            (den[d * heads + hh] + 1e-16f);
        atomicAdd(&out[(size_t)d * HD + f], h[(size_t)s * HD + f] * alpha);
    }
}

// bias (+ optional BN) + ELU; result written as f16 (next GEMM's A matrix).
// mode 0: bias+BN+ELU   mode 1: bias+ELU
__global__ void bias_bn_elu_f16(const float* __restrict__ x, const float* __restrict__ bias,
                                const float* __restrict__ g, const float* __restrict__ be,
                                const float* __restrict__ m, const float* __restrict__ v,
                                _Float16* __restrict__ out16,
                                int total, int C, int mode)
{
    const int i = blockIdx.x * blockDim.x + threadIdx.x;
    if (i >= total) return;
    const int c = i % C;
    float val = x[i] + bias[c];
    if (mode == 0)
        val = (val - m[c]) * rsqrtf(v[c] + 1e-5f) * g[c] + be[c];
    val = val > 0.0f ? val : (__expf(val) - 1.0f);   // ELU(alpha=1)
    out16[i] = (_Float16)val;
}

// tiny final layer: out[n,c] = b[c] + <h[n,:], W[c,:]>   (C = 6, K = 128)
__global__ void fc_small(const float* __restrict__ h, const float* __restrict__ W,
                         const float* __restrict__ b, float* __restrict__ out,
                         int Nn, int K, int C)
{
    const int idx = blockIdx.x * blockDim.x + threadIdx.x;
    if (idx >= Nn * C) return;
    const int n = idx / C, c = idx % C;
    const float* __restrict__ hp = h + (size_t)n * K;
    const float* __restrict__ wp = W + (size_t)c * K;
    float s = b[c];
    for (int k = 0; k < K; ++k) s += hp[k] * wp[k];
    out[idx] = s;
}

// ---------------------------------------------------------------------------
// host-side orchestration
// ---------------------------------------------------------------------------
extern "C" void kernel_launch(void* const* d_in, const int* in_sizes, int n_in,
                              void* d_out, int out_size, void* d_ws, size_t ws_size,
                              hipStream_t stream)
{
    const float* x   = (const float*)d_in[0];
    const int*   ei  = (const int*)  d_in[1];
    const float* W1  = (const float*)d_in[2];
    const float* a1s = (const float*)d_in[3];
    const float* a1d = (const float*)d_in[4];
    const float* b1  = (const float*)d_in[5];
    const float* W2  = (const float*)d_in[6];
    const float* a2s = (const float*)d_in[7];
    const float* a2d = (const float*)d_in[8];
    const float* b2  = (const float*)d_in[9];
    const float* W3  = (const float*)d_in[10];
    const float* a3s = (const float*)d_in[11];
    const float* a3d = (const float*)d_in[12];
    const float* b3  = (const float*)d_in[13];
    const float* g1  = (const float*)d_in[14];
    const float* be1 = (const float*)d_in[15];
    const float* m1  = (const float*)d_in[16];
    const float* v1  = (const float*)d_in[17];
    const float* g2  = (const float*)d_in[18];
    const float* be2 = (const float*)d_in[19];
    const float* m2  = (const float*)d_in[20];
    const float* v2  = (const float*)d_in[21];
    const float* fc1W = (const float*)d_in[22];
    const float* fc1b = (const float*)d_in[23];
    const float* fc2W = (const float*)d_in[24];
    const float* fc2b = (const float*)d_in[25];

    const int N   = in_sizes[0] / F_IN;      // 10000
    const int E   = in_sizes[1] / 2;         // 160000
    const int HD  = HEADS * DIM;             // 1024
    const int tot = E + N;                   // edges incl. self loops

    // ---- workspace layout ---------------------------------------------
    float* wsf = (float*)d_ws;
    size_t off = 0;
    float*    h_cur = wsf + off;  off += (size_t)N * HD;       // GEMM outputs (f32)
    float*    h_agg = wsf + off;  off += (size_t)N * HD;       // aggregation out (f32)
    float*    as    = wsf + off;  off += (size_t)N * HEADS;
    float*    ad    = wsf + off;  off += (size_t)N * HEADS;
    unsigned* emax  = (unsigned*)(wsf + off); off += (size_t)N * HEADS;
    float*    den   = wsf + off;  off += (size_t)N * HEADS;
    float*    ebuf  = wsf + off;  off += (size_t)tot * HEADS;
    float*    fc1o  = wsf + off;  off += (size_t)N * 128;
    _Float16* hbase = (_Float16*)(wsf + off);
    size_t hoff = 0;
    _Float16* xh    = hbase + hoff;  hoff += (size_t)N * F_IN_PAD;     // padded input
    _Float16* h16   = hbase + hoff;  hoff += (size_t)N * HD;           // activations f16
    _Float16* W1h   = hbase + hoff;  hoff += (size_t)HD * F_IN_PAD;
    _Float16* W2h   = hbase + hoff;  hoff += (size_t)HD * HD;
    _Float16* W3h   = hbase + hoff;  hoff += (size_t)DIM * HD;
    _Float16* fc1Wh = hbase + hoff;  hoff += (size_t)128 * DIM;
    (void)ws_size; (void)n_in;

    const int mblocks = (N + 127) / 128;
    auto cvt = [&](const float* s, _Float16* dt, int rows, int K, int Kpad) {
        const size_t n = (size_t)rows * Kpad;
        cvt_f16_pad<<<(unsigned)((n + 255) / 256), 256, 0, stream>>>(s, dt, rows, K, Kpad);
    };

    // ---- one-time per-launch f16 conversions (weights + input) --------
    cvt(x,    xh,    N,   F_IN, F_IN_PAD);
    cvt(W1,   W1h,   HD,  F_IN, F_IN_PAD);
    cvt(W2,   W2h,   HD,  HD,   HD);
    cvt(W3,   W3h,   DIM, HD,   HD);
    cvt(fc1W, fc1Wh, 128, DIM,  DIM);

    // generic GAT layer: Ah(f16) --GEMM(Wh)--> h_cur(f32) --attn/agg--> h_agg
    auto gat_layer = [&](const _Float16* Ah, int Kpad, const _Float16* Wh,
                         const float* asrc, const float* adst, int heads)
    {
        const int hd = heads * DIM;
        wmma_gemm_f16<<<dim3(mblocks, hd / 64), 256, 0, stream>>>(
            Ah, Wh, nullptr, h_cur, N, hd, Kpad, 0);
        alpha_kernel<<<dim3(N, heads), 32, 0, stream>>>(
            h_cur, asrc, adst, as, ad, heads, DIM);
        const size_t nh = (size_t)N * heads;
        fill_u32<<<(unsigned)((nh + 255) / 256), 256, 0, stream>>>(emax, NEG_INF_ORD, nh);
        fill_u32<<<(unsigned)((nh + 255) / 256), 256, 0, stream>>>((unsigned*)den, 0u, nh);
        const size_t no = (size_t)N * hd;
        fill_u32<<<(unsigned)((no + 255) / 256), 256, 0, stream>>>((unsigned*)h_agg, 0u, no);
        edge_logits<<<(tot + 255) / 256, 256, 0, stream>>>(
            ei, E, tot, heads, as, ad, ebuf, emax);
        edge_softmax_num<<<(tot + 255) / 256, 256, 0, stream>>>(
            ei, E, tot, heads, emax, ebuf, den);
        edge_scatter<<<tot, 256, 0, stream>>>(
            ei, E, heads, DIM, h_cur, ebuf, den, h_agg);
    };

    // ---- layer 1: F_IN -> H*D, + BN1 + ELU -> h16 ---------------------
    gat_layer(xh, F_IN_PAD, W1h, a1s, a1d, HEADS);
    bias_bn_elu_f16<<<(N * HD + 255) / 256, 256, 0, stream>>>(
        h_agg, b1, g1, be1, m1, v1, h16, N * HD, HD, 0);

    // ---- layer 2: H*D -> H*D, + BN2 + ELU -> h16 ----------------------
    gat_layer(h16, HD, W2h, a2s, a2d, HEADS);
    bias_bn_elu_f16<<<(N * HD + 255) / 256, 256, 0, stream>>>(
        h_agg, b2, g2, be2, m2, v2, h16, N * HD, HD, 0);

    // ---- layer 3: H*D -> D (heads = 1), + ELU -> h16 ------------------
    gat_layer(h16, HD, W3h, a3s, a3d, 1);
    bias_bn_elu_f16<<<(N * DIM + 255) / 256, 256, 0, stream>>>(
        h_agg, b3, nullptr, nullptr, nullptr, nullptr, h16, N * DIM, DIM, 1);

    // ---- MLP head -----------------------------------------------------
    // fc1: [N,256] x [128,256]^T + b, ReLU  (WMMA)
    wmma_gemm_f16<<<dim3(mblocks, 128 / 64), 256, 0, stream>>>(
        h16, fc1Wh, fc1b, fc1o, N, 128, DIM, 1);
    // fc2: [N,128] x [6,128]^T + b
    fc_small<<<(N * NCLS + 255) / 256, 256, 0, stream>>>(
        fc1o, fc2W, fc2b, (float*)d_out, N, 128, NCLS);
    (void)out_size;
}
